// MovieCnnLayer_13529146982459
// MI455X (gfx1250) — matrix operand order repaired and checked by reading, not compiled, not run on hardware
//
#include <hip/hip_runtime.h>

typedef __attribute__((ext_vector_type(16))) _Float16 v16h;
typedef __attribute__((ext_vector_type(8)))  float    v8f;

namespace {
constexpr int SEQ     = 15;
constexpr int EMBED   = 32;
constexpr int FILTERS = 8;
constexpr int BATCH   = 131072;
constexpr int WAVES_PER_BLOCK = 8;
constexpr int TILES   = BATCH / 16;                 // 16 batch rows per wave
}

// One wave32 handles a 16-row batch tile.
//   A (16x32 f16)  : x[b, s, :] for the tile, one fragment per s, contiguous k = 16h + j layout.
//   B (32x16 f16)  : two conv branches packed into the 16 columns.
//   D (16x16 f32)  : per-(t) conv outputs for 16 rows x 16 filter-columns.
__global__ __launch_bounds__(WAVES_PER_BLOCK * 32)
void textcnn_wmma_kernel(const int*   __restrict__ ids,
                         const float* __restrict__ embed,
                         const float* __restrict__ W0, const float* __restrict__ b0,
                         const float* __restrict__ W1, const float* __restrict__ b1,
                         const float* __restrict__ W2, const float* __restrict__ b2,
                         const float* __restrict__ W3, const float* __restrict__ b3,
                         float* __restrict__ out)
{
  const int lane  = threadIdx.x & 31;
  const int wave  = threadIdx.x >> 5;
  const int tile  = blockIdx.x * WAVES_PER_BLOCK + wave;
  const int bbase = tile * 16;

  const int N = lane & 15;     // matrix column / A row within tile
  const int h = lane >> 4;     // lane half

  // ---------------- A fragments: f16(embed[ids[b,s]]) for every s ----------
  // A layout used here (contiguous k): lanes 0-15 hold K 0..15 (elems 0-15);
  // lanes 16-31 hold K 16..31. Row M = lane & 15.
  const int k0   = 16 * h;
  const int k1   = 16 * h + 8;
  const int arow = bbase + N;
  v16h A[SEQ];
#pragma unroll
  for (int s = 0; s < SEQ; ++s) {
    const int id = ids[arow * SEQ + s];
    const float* e = embed + id * EMBED;
    const float4 p0 = *(const float4*)(e + k0);
    const float4 p1 = *(const float4*)(e + k0 + 4);
    const float4 p2 = *(const float4*)(e + k1);
    const float4 p3 = *(const float4*)(e + k1 + 4);
    v16h a;
    a[0]  = (_Float16)p0.x; a[1]  = (_Float16)p0.y; a[2]  = (_Float16)p0.z; a[3]  = (_Float16)p0.w;
    a[4]  = (_Float16)p1.x; a[5]  = (_Float16)p1.y; a[6]  = (_Float16)p1.z; a[7]  = (_Float16)p1.w;
    a[8]  = (_Float16)p2.x; a[9]  = (_Float16)p2.y; a[10] = (_Float16)p2.z; a[11] = (_Float16)p2.w;
    a[12] = (_Float16)p3.x; a[13] = (_Float16)p3.y; a[14] = (_Float16)p3.z; a[15] = (_Float16)p3.w;
    A[s] = a;
  }

  // ---------------- B fragments ---------------------------------------------
  // B layout used here (contiguous k, as A): lane column = lane&15; lanes 0-15 hold K 0..15,
  // lanes 16-31 hold K 16..31; element j of the fragment = B[Koff + j][N].
  const int  Koff   = 16 * h;
  const int  f      = N & 7;
  const bool lowcol = (N < 8);

  // group 0: window 2 -> cols 0..7, window 3 -> cols 8..15  (3 K-slices)
  v16h Bg0[3];
  {
    const float* Wp  = lowcol ? W0 : W1;
    const int    wln = lowcol ? 2 : 3;
#pragma unroll
    for (int k = 0; k < 3; ++k) {
      v16h b;
#pragma unroll
      for (int j = 0; j < 16; ++j) {
        const float v = (k < wln) ? Wp[((k * EMBED) + (Koff + j)) * FILTERS + f] : 0.0f;
        b[j] = (_Float16)v;
      }
      Bg0[k] = b;
    }
  }
  // group 1: window 4 -> cols 0..7, window 5 -> cols 8..15  (5 K-slices)
  v16h Bg1[5];
  {
    const float* Wp  = lowcol ? W2 : W3;
    const int    wln = lowcol ? 4 : 5;
#pragma unroll
    for (int k = 0; k < 5; ++k) {
      v16h b;
#pragma unroll
      for (int j = 0; j < 16; ++j) {
        const float v = (k < wln) ? Wp[((k * EMBED) + (Koff + j)) * FILTERS + f] : 0.0f;
        b[j] = (_Float16)v;
      }
      Bg1[k] = b;
    }
  }

  const float bias0 = lowcol ? b0[f] : b1[f];
  const float bias1 = lowcol ? b2[f] : b3[f];

  // ---------------- group 0 compute: t = 0..13 ------------------------------
  {
    float mx[8];
#pragma unroll
    for (int r = 0; r < 8; ++r) mx[r] = 0.0f;    // ReLU outputs are >= 0
#pragma unroll
    for (int t = 0; t <= SEQ - 2; ++t) {
      v8f acc = {};
#pragma unroll
      for (int k = 0; k < 3; ++k) {
        if (t + k <= SEQ - 1) {                  // compile-time after unroll
          acc = __builtin_amdgcn_wmma_f32_16x16x32_f16(
              false, A[t + k], false, Bg0[k], (short)0, acc, false, false);
        }
      }
      const bool valid = (t <= SEQ - 3) || lowcol;   // w=3 needs t <= 12
#pragma unroll
      for (int r = 0; r < 8; ++r) {
        float y = acc[r] + bias0;
        y = y > 0.0f ? y : 0.0f;
        mx[r] = valid ? fmaxf(mx[r], y) : mx[r];     // VALU select, no branch
      }
    }
    // D layout: acc[r] is row (r + 8h), col N  ->  out[b, 0, N]
#pragma unroll
    for (int r = 0; r < 8; ++r)
      out[(bbase + r + 8 * h) * 32 + N] = mx[r];
  }

  // ---------------- group 1 compute: t = 0..11 ------------------------------
  {
    float mx[8];
#pragma unroll
    for (int r = 0; r < 8; ++r) mx[r] = 0.0f;
#pragma unroll
    for (int t = 0; t <= SEQ - 4; ++t) {
      v8f acc = {};
#pragma unroll
      for (int k = 0; k < 5; ++k) {
        if (t + k <= SEQ - 1) {
          acc = __builtin_amdgcn_wmma_f32_16x16x32_f16(
              false, A[t + k], false, Bg1[k], (short)0, acc, false, false);
        }
      }
      const bool valid = (t <= SEQ - 5) || lowcol;   // w=5 needs t <= 10
#pragma unroll
      for (int r = 0; r < 8; ++r) {
        float y = acc[r] + bias1;
        y = y > 0.0f ? y : 0.0f;
        mx[r] = valid ? fmaxf(mx[r], y) : mx[r];
      }
    }
#pragma unroll
    for (int r = 0; r < 8; ++r)
      out[(bbase + r + 8 * h) * 32 + 16 + N] = mx[r];
  }
}

extern "C" void kernel_launch(void* const* d_in, const int* in_sizes, int n_in,
                              void* d_out, int out_size, void* d_ws, size_t ws_size,
                              hipStream_t stream) {
  const int*   ids = (const int*)d_in[0];
  const float* emb = (const float*)d_in[1];
  const float* W0  = (const float*)d_in[2]; const float* b0 = (const float*)d_in[3];
  const float* W1  = (const float*)d_in[4]; const float* b1 = (const float*)d_in[5];
  const float* W2  = (const float*)d_in[6]; const float* b2 = (const float*)d_in[7];
  const float* W3  = (const float*)d_in[8]; const float* b3 = (const float*)d_in[9];
  float* out = (float*)d_out;

  dim3 block(WAVES_PER_BLOCK * 32);
  dim3 grid(TILES / WAVES_PER_BLOCK);   // 8192 waves / 8 per block = 1024 blocks
  textcnn_wmma_kernel<<<grid, block, 0, stream>>>(ids, emb, W0, b0, W1, b1,
                                                  W2, b2, W3, b3, out);
}
